// E_02_HSE_v2_10376640987786
// MI455X (gfx1250) — compile-verified, matrix operand order checked by reading, not run on hardware
//
#include <hip/hip_runtime.h>
#include <hip/hip_bf16.h>

typedef __attribute__((ext_vector_type(16))) _Float16 v16h;
typedef __attribute__((ext_vector_type(8)))  float    v8f;

#define B_    32
#define L_    8192
#define C_    8
#define P_    512
#define PL_   64
#define PC_   4
#define CED_  8
#define SED_  16
#define D_    256
#define INDIM 2576
#define KPAD  2592
#define KT1   81      // KPAD / 32
#define KT2   8       // 256 / 32
#define NT    16      // 256 / 16

// WMMA fragment half-index -> K offset mapping (16-bit A/B, 16x16x32)
__device__ __forceinline__ int koff(int lane, int h) {
    int base = (h < 8) ? ((lane < 16) ? 0 : 8) : ((lane < 16) ? 16 : 24);
    return base + (h & 7);
}

union AFrag { v16h v; _Float16 h[16]; };

// ---------------------------------------------------------------------------
// Prep: write g[k]*W[k,n] into WMMA-B fragment order: [kt][nt][lane][16 halves]
// ---------------------------------------------------------------------------
__global__ void prep_wfrag(const float* __restrict__ W, const float* __restrict__ g,
                           _Float16* __restrict__ out, int K, int Ktiles) {
    int gid = blockIdx.x * blockDim.x + threadIdx.x;
    int total = Ktiles * NT * 32;
    if (gid >= total) return;
    int lane = gid & 31;
    int fi   = gid >> 5;
    int kt   = fi / NT, nt = fi % NT;
    int n    = nt * 16 + (lane & 15);
    _Float16* dst = out + (((size_t)(kt * NT + nt)) * 32 + lane) * 16;
#pragma unroll
    for (int h = 0; h < 16; ++h) {
        int k = kt * 32 + koff(lane, h);
        float v = (k < K) ? g[k] * W[(size_t)k * D_ + n] : 0.0f;
        dst[h] = (_Float16)v;
    }
}

// ---------------------------------------------------------------------------
// Prep: epilogue vectors  cs = g@W,  c0 = b_ln@W + bias   (for both layers)
// ---------------------------------------------------------------------------
__global__ void prep_vec(const float* __restrict__ W1, const float* __restrict__ g1,
                         const float* __restrict__ bl1, const float* __restrict__ bias1,
                         const float* __restrict__ W2, const float* __restrict__ g2,
                         const float* __restrict__ bl2, const float* __restrict__ bias2,
                         float* __restrict__ cs1, float* __restrict__ c01,
                         float* __restrict__ cs2, float* __restrict__ c02) {
    int n = threadIdx.x;   // 256 threads, one column each
    float a = 0.f, b = 0.f;
    for (int k = 0; k < INDIM; ++k) {
        float w = W1[(size_t)k * D_ + n];
        a += g1[k] * w;
        b += bl1[k] * w;
    }
    cs1[n] = a; c01[n] = b + bias1[n];
    a = 0.f; b = 0.f;
    for (int k = 0; k < D_; ++k) {
        float w = W2[(size_t)k * D_ + n];
        a += g2[k] * w;
        b += bl2[k] * w;
    }
    cs2[n] = a; c02[n] = b + bias2[n];
}

// ---------------------------------------------------------------------------
// Fused: feature build -> LN1-folded GEMM1 -> SiLU -> LN2-folded GEMM2
// One 256-thread workgroup per 16 output rows. 8 waves x 2 N-tiles each.
// ---------------------------------------------------------------------------
__global__ __launch_bounds__(256) void fused_embed(
    const float* __restrict__ x, const float* __restrict__ fs,
    const float* __restrict__ cemb, const float* __restrict__ semb,
    const int* __restrict__ startL, const int* __restrict__ startC,
    const int* __restrict__ ctm, const int* __restrict__ sysid_p,
    const _Float16* __restrict__ w1f, const _Float16* __restrict__ w2f,
    const float* __restrict__ cs1, const float* __restrict__ c01,
    const float* __restrict__ cs2, const float* __restrict__ c02,
    float* __restrict__ out)
{
    __shared__ _Float16 featAB[16 * 512];   // data+time part (k < 512), f16
    __shared__ _Float16 ceLDS[16 * 32];     // 32 channel-embed values per row
    __shared__ _Float16 sys16[SED_];
    __shared__ float    htile[16 * 264];    // silu(h1), fp32, padded stride
    __shared__ float    psum[256], pssq[256];
    __shared__ int      sL_sh[16], sC_sh[16], b_sh[16], ct_sh[C_];
    __shared__ float    invfs_sh[16], mu_sh[16], rstd_sh[16];

    const int tid  = threadIdx.x;
    const int lane = tid & 31;
    const int wv   = tid >> 5;
    const int r0   = blockIdx.x * 16;
    const int sid  = *sysid_p;

    // ---- Stage 1: per-row parameters + small tables -----------------------
    if (tid < 16) {
        int r = r0 + tid;
        int b = r >> 9, p = r & (P_ - 1);
        sL_sh[tid]    = startL[b * P_ + p];
        sC_sh[tid]    = startC[b * P_ + p];
        invfs_sh[tid] = 1.0f / fs[b];
        b_sh[tid]     = b;
    } else if (tid < 16 + C_) {
        ct_sh[tid - 16] = ctm[sid * 20 + (tid - 16)];
    } else if (tid < 16 + C_ + SED_) {
        sys16[tid - 16 - C_] = (_Float16)semb[sid * SED_ + (tid - 16 - C_)];
    }
    __syncthreads();

    // ---- Stage 2: build feature staging + partial stats (16 thr/row) -----
    {
        int r = tid >> 4, q = tid & 15;
        int sL = sL_sh[r], sC = sC_sh[r], b = b_sh[r];
        float invfs = invfs_sh[r];
        float s = 0.f, ss = 0.f;
        if (q < 8) {                       // data_f (k<256) + time_f (256..511)
            int k0 = q * 64;
            for (int k = k0; k < k0 + 64; ++k) {
                float v;
                if (k < 256) {
                    int i  = k >> 2, j = k & 3;
                    int li = (sL + i) & (L_ - 1);
                    int ci = (sC + j) & (C_ - 1);
                    v = x[((size_t)b * L_ + li) * C_ + ci];
                } else {
                    int i  = (k - 256) >> 2;
                    int li = (sL + i) & (L_ - 1);
                    v = (float)li * invfs;
                }
                _Float16 hv = (_Float16)v;
                featAB[r * 512 + k] = hv;
                float vf = (float)hv;
                s += vf; ss += vf * vf;
            }
        } else if (q < 12) {               // 32 channel-embed values (x64 bcast)
            int i0 = (q - 8) * 8;
            for (int idx = i0; idx < i0 + 8; ++idx) {
                int j = idx >> 3, e = idx & 7;
                int ci = (sC + j) & (C_ - 1);
                float v = cemb[ct_sh[ci] * CED_ + e];
                _Float16 hv = (_Float16)v;
                ceLDS[r * 32 + idx] = hv;
                float vf = (float)hv;
                s += 64.f * vf; ss += 64.f * vf * vf;
            }
        } else if (q == 12) {              // system embed contribution
            for (int e = 0; e < SED_; ++e) {
                float vf = (float)sys16[e];
                s += vf; ss += vf * vf;
            }
        }
        psum[tid] = s; pssq[tid] = ss;
    }
    __syncthreads();

    // ---- Stage 3: LN1 row stats ------------------------------------------
    if (tid < 16) {
        float s = 0.f, ss = 0.f;
        for (int q = 0; q < 16; ++q) { s += psum[tid * 16 + q]; ss += pssq[tid * 16 + q]; }
        float mu  = s / (float)INDIM;
        float var = ss / (float)INDIM - mu * mu;
        mu_sh[tid]   = mu;
        rstd_sh[tid] = rsqrtf(var + 1e-5f);
    }
    __syncthreads();

    // ---- Stage 4: GEMM1 (K = 2592 in 81 tiles) ---------------------------
    const int m    = lane & 15;
    const int kblo = (lane < 16) ? 0  : 8;
    const int kbhi = (lane < 16) ? 16 : 24;
    v8f acc0 = {}, acc1 = {};

    for (int kt = 0; kt < 16; ++kt) {                // data+time tiles from LDS
        AFrag a;
#pragma unroll
        for (int h = 0; h < 8; ++h) a.h[h]     = featAB[m * 512 + kt * 32 + kblo + h];
#pragma unroll
        for (int h = 0; h < 8; ++h) a.h[8 + h] = featAB[m * 512 + kt * 32 + kbhi + h];
        const _Float16* bp = w1f + (((size_t)(kt * NT + 2 * wv)) * 32 + lane) * 16;
        v16h bf0 = *(const v16h*)bp;
        v16h bf1 = *(const v16h*)(bp + 32 * 16);
        acc0 = __builtin_amdgcn_wmma_f32_16x16x32_f16(false, a.v, false, bf0, (short)0, acc0, false, false);
        acc1 = __builtin_amdgcn_wmma_f32_16x16x32_f16(false, a.v, false, bf1, (short)0, acc1, false, false);
    }

    AFrag ace;                                       // identical A for kt 16..79
#pragma unroll
    for (int h = 0; h < 8; ++h) ace.h[h]     = ceLDS[m * 32 + kblo + h];
#pragma unroll
    for (int h = 0; h < 8; ++h) ace.h[8 + h] = ceLDS[m * 32 + kbhi + h];
    for (int kt = 16; kt < 80; ++kt) {
        const _Float16* bp = w1f + (((size_t)(kt * NT + 2 * wv)) * 32 + lane) * 16;
        v16h bf0 = *(const v16h*)bp;
        v16h bf1 = *(const v16h*)(bp + 32 * 16);
        acc0 = __builtin_amdgcn_wmma_f32_16x16x32_f16(false, ace.v, false, bf0, (short)0, acc0, false, false);
        acc1 = __builtin_amdgcn_wmma_f32_16x16x32_f16(false, ace.v, false, bf1, (short)0, acc1, false, false);
    }

    {                                                // kt == 80: system + pad
        AFrag asy;
#pragma unroll
        for (int h = 0; h < 8; ++h) { asy.h[h] = sys16[kblo + h]; asy.h[8 + h] = (_Float16)0.f; }
        const _Float16* bp = w1f + (((size_t)(80 * NT + 2 * wv)) * 32 + lane) * 16;
        v16h bf0 = *(const v16h*)bp;
        v16h bf1 = *(const v16h*)(bp + 32 * 16);
        acc0 = __builtin_amdgcn_wmma_f32_16x16x32_f16(false, asy.v, false, bf0, (short)0, acc0, false, false);
        acc1 = __builtin_amdgcn_wmma_f32_16x16x32_f16(false, asy.v, false, bf1, (short)0, acc1, false, false);
    }

    // ---- Stage 5: LN1 epilogue + SiLU -> htile ---------------------------
    {
        int n0 = 32 * wv + (lane & 15);
        int n1 = n0 + 16;
        float a0 = cs1[n0], a1 = cs1[n1];
        float d0 = c01[n0], d1 = c01[n1];
        int mb = (lane < 16) ? 0 : 8;
#pragma unroll
        for (int v = 0; v < 8; ++v) {
            int mm = v + mb;
            float rs = rstd_sh[mm], mu = mu_sh[mm];
            float h1 = rs * acc0[v] - rs * mu * a0 + d0;
            htile[mm * 264 + n0] = h1 / (1.f + __expf(-h1));
            h1 = rs * acc1[v] - rs * mu * a1 + d1;
            htile[mm * 264 + n1] = h1 / (1.f + __expf(-h1));
        }
    }
    __syncthreads();

    // ---- Stage 6: LN2 row stats ------------------------------------------
    {
        int r = tid >> 4, q = tid & 15;
        float s = 0.f, ss = 0.f;
        for (int i = 0; i < 16; ++i) {
            float vf = htile[r * 264 + q * 16 + i];
            s += vf; ss += vf * vf;
        }
        psum[tid] = s; pssq[tid] = ss;
    }
    __syncthreads();
    if (tid < 16) {
        float s = 0.f, ss = 0.f;
        for (int q = 0; q < 16; ++q) { s += psum[tid * 16 + q]; ss += pssq[tid * 16 + q]; }
        float mu  = s / (float)D_;
        float var = ss / (float)D_ - mu * mu;
        mu_sh[tid]   = mu;
        rstd_sh[tid] = rsqrtf(var + 1e-5f);
    }
    __syncthreads();

    // ---- Stage 7: GEMM2 (K = 256 in 8 tiles) -----------------------------
    v8f acc20 = {}, acc21 = {};
    for (int kt = 0; kt < KT2; ++kt) {
        AFrag a;
#pragma unroll
        for (int h = 0; h < 8; ++h) a.h[h]     = (_Float16)htile[m * 264 + kt * 32 + kblo + h];
#pragma unroll
        for (int h = 0; h < 8; ++h) a.h[8 + h] = (_Float16)htile[m * 264 + kt * 32 + kbhi + h];
        const _Float16* bp = w2f + (((size_t)(kt * NT + 2 * wv)) * 32 + lane) * 16;
        v16h bf0 = *(const v16h*)bp;
        v16h bf1 = *(const v16h*)(bp + 32 * 16);
        acc20 = __builtin_amdgcn_wmma_f32_16x16x32_f16(false, a.v, false, bf0, (short)0, acc20, false, false);
        acc21 = __builtin_amdgcn_wmma_f32_16x16x32_f16(false, a.v, false, bf1, (short)0, acc21, false, false);
    }

    // ---- Stage 8: LN2 epilogue + store -----------------------------------
    {
        int n0 = 32 * wv + (lane & 15);
        int n1 = n0 + 16;
        float a0 = cs2[n0], a1 = cs2[n1];
        float d0 = c02[n0], d1 = c02[n1];
        int mb = (lane < 16) ? 0 : 8;
#pragma unroll
        for (int v = 0; v < 8; ++v) {
            int mm = v + mb;
            float rs = rstd_sh[mm], mu = mu_sh[mm];
            out[(size_t)(r0 + mm) * D_ + n0] = rs * acc20[v] - rs * mu * a0 + d0;
            out[(size_t)(r0 + mm) * D_ + n1] = rs * acc21[v] - rs * mu * a1 + d1;
        }
    }
}

// ---------------------------------------------------------------------------
extern "C" void kernel_launch(void* const* d_in, const int* in_sizes, int n_in,
                              void* d_out, int out_size, void* d_ws, size_t ws_size,
                              hipStream_t stream) {
    const float* x     = (const float*)d_in[0];
    const float* fs    = (const float*)d_in[1];
    const float* cemb  = (const float*)d_in[2];
    const float* semb  = (const float*)d_in[3];
    const float* ln1_g = (const float*)d_in[4];
    const float* ln1_b = (const float*)d_in[5];
    const float* W1    = (const float*)d_in[6];
    const float* b1    = (const float*)d_in[7];
    const float* ln2_g = (const float*)d_in[8];
    const float* ln2_b = (const float*)d_in[9];
    const float* W2    = (const float*)d_in[10];
    const float* b2    = (const float*)d_in[11];
    const int* startL  = (const int*)d_in[12];
    const int* startC  = (const int*)d_in[13];
    const int* ctm     = (const int*)d_in[14];
    const int* sysid   = (const int*)d_in[15];

    // workspace layout
    const size_t W1F_BYTES = (size_t)KT1 * NT * 32 * 16 * sizeof(_Float16); // 1,327,104
    const size_t W2F_BYTES = (size_t)KT2 * NT * 32 * 16 * sizeof(_Float16); //   131,072
    char* ws = (char*)d_ws;
    _Float16* w1f = (_Float16*)ws;
    _Float16* w2f = (_Float16*)(ws + W1F_BYTES);
    float* cs1 = (float*)(ws + W1F_BYTES + W2F_BYTES);
    float* c01 = cs1 + D_;
    float* cs2 = c01 + D_;
    float* c02 = cs2 + D_;

    // prep: swizzle g1*W1 and g2*W2 into WMMA B-fragment order (f16)
    prep_wfrag<<<(KT1 * NT * 32 + 255) / 256, 256, 0, stream>>>(W1, ln1_g, w1f, INDIM, KT1);
    prep_wfrag<<<(KT2 * NT * 32 + 255) / 256, 256, 0, stream>>>(W2, ln2_g, w2f, D_, KT2);
    prep_vec<<<1, 256, 0, stream>>>(W1, ln1_g, ln1_b, b1, W2, ln2_g, ln2_b, b2,
                                    cs1, c01, cs2, c02);

    // fused main kernel: 16384 rows / 16 rows per workgroup
    fused_embed<<<(B_ * P_) / 16, 256, 0, stream>>>(
        x, fs, cemb, semb, startL, startC, ctm, sysid,
        w1f, w2f, cs1, c01, cs2, c02, (float*)d_out);
}
